// WDiscriminatorMask_11029476016801
// MI455X (gfx1250) — matrix operand order repaired and checked by reading, not compiled, not run on hardware
//
#include <hip/hip_runtime.h>
#include <hip/hip_bf16.h>

#define SLOPE 0.2f
#define EPSV 1e-5f

typedef __attribute__((ext_vector_type(16))) __bf16 bf16x16;
typedef __attribute__((ext_vector_type(8)))  float  f32x8;
typedef __attribute__((ext_vector_type(4)))  unsigned int u32x4;

union FragAB {
    u32x4        q[2];
    unsigned int u[8];
    bf16x16      v;
};

__device__ __forceinline__ unsigned int f2bf(float f) {
    unsigned int u = __float_as_uint(f);
    return (u + 0x7FFFu + ((u >> 16) & 1u)) >> 16;   // RNE bf16
}
__device__ __forceinline__ float bf2f(unsigned int h) {
    return __uint_as_float((h & 0xFFFFu) << 16);
}

// ---------------------------------------------------------------------------
// Implicit-GEMM 3x3 VALID conv, bf16 WMMA, f32 accum, NHWC pair-packed.
// in  : bf16 [Hin*Win][Cin/2] dwords (even channel in lo16)
// wpk : bf16 prepacked weights [9][Cin/32][Cout][32 bf16 = 16 dwords]
// y   : bf16 [Hout*Wout][Cout/2] dwords
// Wave = 32 out-channels x 16 pixels (2 accumulators, shared B fragment).
// WG = 8 waves = 32co x 128px tile. B loads: 2x b128; A loads: 4x b128.
// ---------------------------------------------------------------------------
__global__ __launch_bounds__(256) void conv3x3_wmma_bf16(
    const unsigned int* __restrict__ in,
    const unsigned int* __restrict__ wpk,
    unsigned int* __restrict__ y,
    int Cin, int Cout, int Hin, int Win)
{
    const int Hout = Hin - 2, Wout = Win - 2;
    const int lane = threadIdx.x & 31;
    const int wave = threadIdx.x >> 5;        // pixel group 0..7
    const int n    = lane & 15;               // pixel column / A row
    const int half = lane >> 4;               // K-half selector

    const int co0 = blockIdx.z * 32;
    const int row = blockIdx.y;
    const int xo  = blockIdx.x * 128 + wave * 16 + n;
    const int xc  = (xo < Wout) ? xo : (Wout - 1);

    const int Cpi    = Cin >> 1;              // input channel pairs
    const int nchunk = Cin >> 5;

    f32x8 acc0 = {}, acc1 = {};

    for (int t = 0; t < 9; ++t) {
        const int kr = t / 3, kc = t - kr * 3;
        // B: 16-bit table layout -> cpairs ch*16 + [half*4..+3] and [8+half*4..+3]
        // -> two b128 loads.
        const unsigned int* bbase =
            in + ((size_t)(row + kr) * Win + (xc + kc)) * Cpi + half * 4;
        // A: 16-bit A layout -> dwords [half*4..+3] and [8+half*4..+3]
        const unsigned int* abase =
            wpk + ((size_t)(t * nchunk) * Cout + co0 + n) * 16 + half * 4;
        for (int ch = 0; ch < nchunk; ++ch) {
            FragAB a0, a1, b;
            const unsigned int* ap = abase + (size_t)ch * Cout * 16;
            a0.q[0] = *(const u32x4*)(ap);
            a0.q[1] = *(const u32x4*)(ap + 8);
            a1.q[0] = *(const u32x4*)(ap + 256);   // +16 channels * 16 dwords
            a1.q[1] = *(const u32x4*)(ap + 264);
            const unsigned int* bp = bbase + ch * 16;
            b.q[0] = *(const u32x4*)(bp);
            b.q[1] = *(const u32x4*)(bp + 8);
            acc0 = __builtin_amdgcn_wmma_f32_16x16x32_bf16(
                false, a0.v, false, b.v, (short)0, acc0, false, false);
            acc1 = __builtin_amdgcn_wmma_f32_16x16x32_bf16(
                false, a1.v, false, b.v, (short)0, acc1, false, false);
        }
    }

    if (xo < Wout) {
        const int Cpo = Cout >> 1;
        // D layout: VGPR v -> co = co0 + half*8 + v; pairs are consecutive cpairs
        unsigned int* yp =
            y + ((size_t)row * Wout + xo) * Cpo + (co0 >> 1) + half * 4;
        u32x4 s0, s1;
        #pragma unroll
        for (int j = 0; j < 4; ++j) {
            s0[j] = f2bf(acc0[2 * j]) | (f2bf(acc0[2 * j + 1]) << 16);
            s1[j] = f2bf(acc1[2 * j]) | (f2bf(acc1[2 * j + 1]) << 16);
        }
        *(u32x4*)(yp)     = s0;
        *(u32x4*)(yp + 8) = s1;   // acc1 covers co0+16.. -> cpair offset +8
    }
}

// ---------------------------------------------------------------------------
// Mask dilation: m=1 iff every mask[0,0] pixel in (2r+1)^2 window equals 1.
// Also accumulates count of ones (block reduce + 1 atomic).
// ---------------------------------------------------------------------------
__global__ __launch_bounds__(256) void dilate_mask_kernel(
    const float* __restrict__ mask, float* __restrict__ mout,
    float* __restrict__ count, int Hout, int Wout, int Wm, int rad)
{
    __shared__ float red[256];
    const int x  = blockIdx.x * 16 + (threadIdx.x & 15);
    const int yy = blockIdx.y * 16 + (threadIdx.x >> 4);
    float m = 0.f;
    if (x < Wout && yy < Hout) {
        m = 1.f;
        for (int dy = 0; dy <= 2 * rad; ++dy)
            for (int dx = 0; dx <= 2 * rad; ++dx)
                if (mask[(yy + dy) * Wm + (x + dx)] != 1.f) m = 0.f;
        mout[yy * Wout + x] = m;
    }
    red[threadIdx.x] = m;
    __syncthreads();
    for (int s = 128; s > 0; s >>= 1) {
        if (threadIdx.x < s) red[threadIdx.x] += red[threadIdx.x + s];
        __syncthreads();
    }
    if (threadIdx.x == 0) atomicAdd(count, red[0]);
}

// ---------------------------------------------------------------------------
// Masked per-channel sum / sumsq over NHWC bf16 y.
// thread -> cpair = tid & (Cpairs-1); replicas over pixels; LDS tree reduce.
// ---------------------------------------------------------------------------
__global__ __launch_bounds__(256) void stats_kernel(
    const unsigned int* __restrict__ y, const float* __restrict__ m,
    float* __restrict__ sum, float* __restrict__ sq, int HW, int lgC)
{
    __shared__ float r0[256], r1[256], r2[256], r3[256];
    const int Cpairs = 1 << lgC;
    const int cp   = threadIdx.x & (Cpairs - 1);
    const int rep  = threadIdx.x >> lgC;
    const int nrep = 256 >> lgC;
    float s0 = 0.f, s1 = 0.f, q0 = 0.f, q1 = 0.f;
    const int base = blockIdx.x * 4096;
    int end = base + 4096; if (end > HW) end = HW;
    for (int p = base + rep; p < end; p += nrep) {
        unsigned int d = y[(size_t)p << lgC | cp];
        float mm = m[p];
        float v0 = bf2f(d) * mm;
        float v1 = bf2f(d >> 16) * mm;
        s0 += v0; s1 += v1; q0 += v0 * v0; q1 += v1 * v1;
    }
    r0[threadIdx.x] = s0; r1[threadIdx.x] = s1;
    r2[threadIdx.x] = q0; r3[threadIdx.x] = q1;
    __syncthreads();
    for (int s = 128; s >= Cpairs; s >>= 1) {
        if (threadIdx.x < s) {
            r0[threadIdx.x] += r0[threadIdx.x + s];
            r1[threadIdx.x] += r1[threadIdx.x + s];
            r2[threadIdx.x] += r2[threadIdx.x + s];
            r3[threadIdx.x] += r3[threadIdx.x + s];
        }
        __syncthreads();
    }
    if (threadIdx.x < Cpairs) {
        atomicAdd(&sum[2 * cp],     r0[threadIdx.x]);
        atomicAdd(&sum[2 * cp + 1], r1[threadIdx.x]);
        atomicAdd(&sq[2 * cp],      r2[threadIdx.x]);
        atomicAdd(&sq[2 * cp + 1],  r3[threadIdx.x]);
    }
}

__global__ void finalize_kernel(const float* __restrict__ sum, const float* __restrict__ sq,
                                const float* __restrict__ count,
                                const float* __restrict__ g, const float* __restrict__ be,
                                float* __restrict__ scale, float* __restrict__ shift, int C)
{
    int c = blockIdx.x * blockDim.x + threadIdx.x;
    if (c < C) {
        float nf = count[0];                 // batch == 1
        float mean = sum[c] / nf;
        float var  = sq[c] / nf - mean * mean;
        float s = g[c] * rsqrtf(var + EPSV);
        scale[c] = s;
        shift[c] = be[c] - mean * s;
    }
}

// In-place NHWC: y = leaky((scale*y + shift) * m)
__global__ __launch_bounds__(256) void norm_kernel(
    unsigned int* __restrict__ y, const float* __restrict__ m,
    const float* __restrict__ scale, const float* __restrict__ shift,
    int lgC, int total)
{
    int idx = blockIdx.x * 256 + threadIdx.x;
    if (idx >= total) return;
    int cp = idx & ((1 << lgC) - 1);
    int p  = idx >> lgC;
    unsigned int d = y[idx];
    float mm = m[p];
    float v0 = (scale[2 * cp]     * bf2f(d)       + shift[2 * cp])     * mm;
    float v1 = (scale[2 * cp + 1] * bf2f(d >> 16) + shift[2 * cp + 1]) * mm;
    v0 = v0 > 0.f ? v0 : v0 * SLOPE;
    v1 = v1 > 0.f ? v1 : v1 * SLOPE;
    y[idx] = f2bf(v0) | (f2bf(v1) << 16);
}

// fp32 weights [Cout][CinReal][3][3] -> bf16 packed [9][CinPad/32][Cout][16 dw]
__global__ void prep_weights_kernel(const float* __restrict__ w, unsigned int* __restrict__ wpk,
                                    int CinReal, int CinPad, int Cout)
{
    const int nchunk = CinPad >> 5;
    int idx = blockIdx.x * 256 + threadIdx.x;
    int total = 9 * nchunk * Cout;
    if (idx >= total) return;
    int co = idx % Cout;
    int t2 = idx / Cout;
    int ch = t2 % nchunk;
    int t  = t2 / nchunk;
    int kr = t / 3, kc = t - kr * 3;
    unsigned int* out = wpk + (size_t)idx * 16;
    for (int j = 0; j < 16; ++j) {
        int ci0 = ch * 32 + 2 * j, ci1 = ci0 + 1;
        float f0 = (ci0 < CinReal) ? w[((size_t)co * CinReal + ci0) * 9 + kr * 3 + kc] : 0.f;
        float f1 = (ci1 < CinReal) ? w[((size_t)co * CinReal + ci1) * 9 + kr * 3 + kc] : 0.f;
        out[j] = f2bf(f0) | (f2bf(f1) << 16);
    }
}

// fp32 x [3][768^2] -> NHWC bf16 pair-packed, zero-padded to 32 channels
__global__ void pack_input_kernel(const float* __restrict__ x, unsigned int* __restrict__ out, int HW)
{
    int p = blockIdx.x * 256 + threadIdx.x;
    if (p >= HW) return;
    u32x4 v0 = {}, vz = {};
    v0[0] = f2bf(x[p]) | (f2bf(x[HW + p]) << 16);
    v0[1] = f2bf(x[2 * HW + p]);
    u32x4* o = (u32x4*)(out + (size_t)p * 16);
    o[0] = v0; o[1] = vz; o[2] = vz; o[3] = vz;
}

// Tail 32->1 conv (tiny, 0.33 GFLOP): direct fp32 MACs from NHWC bf16, * mask
__global__ __launch_bounds__(256) void tail_conv_kernel(
    const unsigned int* __restrict__ in, const float* __restrict__ wt,
    const float* __restrict__ bt, const float* __restrict__ m5,
    float* __restrict__ out, int Hin, int Win)
{
    const int Hout = Hin - 2, Wout = Win - 2;
    const int x = blockIdx.x * 16 + (threadIdx.x & 15);
    const int y = blockIdx.y * 16 + (threadIdx.x >> 4);
    if (x >= Wout || y >= Hout) return;
    float acc = bt[0];
    for (int dy = 0; dy < 3; ++dy)
        for (int dx = 0; dx < 3; ++dx) {
            const unsigned int* ip = in + ((size_t)(y + dy) * Win + (x + dx)) * 16;
            for (int cp = 0; cp < 16; ++cp) {
                unsigned int d = ip[cp];
                acc += bf2f(d)       * wt[(2 * cp) * 9     + dy * 3 + dx];
                acc += bf2f(d >> 16) * wt[(2 * cp + 1) * 9 + dy * 3 + dx];
            }
        }
    out[y * Wout + x] = acc * m5[y * Wout + x];
}

__global__ void zero_kernel(float* __restrict__ p, int n)
{
    int i = blockIdx.x * 256 + threadIdx.x;
    if (i < n) p[i] = 0.f;
}

// ---------------------------------------------------------------------------
extern "C" void kernel_launch(void* const* d_in, const int* in_sizes, int n_in,
                              void* d_out, int out_size, void* d_ws, size_t ws_size,
                              hipStream_t stream)
{
    (void)in_sizes; (void)n_in; (void)out_size; (void)ws_size;

    const float* x    = (const float*)d_in[0];
    const float* mask = (const float*)d_in[1];
    const float* w[4]  = { (const float*)d_in[2],  (const float*)d_in[6],
                           (const float*)d_in[10], (const float*)d_in[14] };
    const float* g[4]  = { (const float*)d_in[4],  (const float*)d_in[8],
                           (const float*)d_in[12], (const float*)d_in[16] };
    const float* be[4] = { (const float*)d_in[5],  (const float*)d_in[9],
                           (const float*)d_in[13], (const float*)d_in[17] };
    const float* wt = (const float*)d_in[18];
    const float* bt = (const float*)d_in[19];
    float* outp = (float*)d_out;

    // Layer geometry
    const int Hin[5]     = {768, 766, 764, 762, 760};
    const int CinReal[4] = {3, 256, 128, 64};
    const int CinPad[4]  = {32, 256, 128, 64};
    const int Cout[4]    = {256, 128, 64, 32};
    const int lgC[4]     = {7, 6, 5, 4};         // log2(Cout/2)
    const int statOff[4] = {0, 256, 384, 448};

    // Workspace carve-up
    size_t off = 0;
    auto carve = [&](size_t bytes) -> char* {
        char* p = (char*)d_ws + off;
        off += (bytes + 255) & ~(size_t)255;
        return p;
    };
    unsigned int* bufA = (unsigned int*)carve((size_t)37356544 * 4);   // 149.4 MB
    unsigned int* bufB = (unsigned int*)carve((size_t)75104768 * 4);   // 300.4 MB
    float* mbuf[5];
    for (int i = 0; i < 5; ++i) {
        int ho = Hin[i] - 2;
        mbuf[i] = (float*)carve((size_t)ho * ho * 4);
    }
    float* counts = (float*)carve(8 * 4);
    float* sum    = (float*)carve(480 * 4);
    float* sq     = (float*)carve(480 * 4);
    float* scale  = (float*)carve(480 * 4);
    float* shift  = (float*)carve(480 * 4);
    unsigned int* wpk[4];
    for (int i = 0; i < 4; ++i) {
        size_t dw = (size_t)9 * (CinPad[i] >> 5) * Cout[i] * 16;
        wpk[i] = (unsigned int*)carve(dw * 4);
    }

    // 1. zero stats + counts (ws is not re-poisoned between replays)
    zero_kernel<<<dim3(4), dim3(256), 0, stream>>>(counts, 8);
    zero_kernel<<<dim3(4), dim3(256), 0, stream>>>(sum, 480);
    zero_kernel<<<dim3(4), dim3(256), 0, stream>>>(sq, 480);

    // 2. prepack weights
    for (int i = 0; i < 4; ++i) {
        int total = 9 * (CinPad[i] >> 5) * Cout[i];
        prep_weights_kernel<<<dim3((total + 255) / 256), dim3(256), 0, stream>>>(
            w[i], wpk[i], CinReal[i], CinPad[i], Cout[i]);
    }

    // 3. pack input x -> bufA (NHWC bf16, 32ch zero-padded)
    {
        int HW = 768 * 768;
        pack_input_kernel<<<dim3((HW + 255) / 256), dim3(256), 0, stream>>>(x, bufA, HW);
    }

    // 4. dilated masks (radii 1..5)
    for (int i = 0; i < 5; ++i) {
        int ho = Hin[i] - 2;
        dilate_mask_kernel<<<dim3((ho + 15) / 16, (ho + 15) / 16), dim3(256), 0, stream>>>(
            mask, mbuf[i], &counts[i], ho, ho, 768, i + 1);
    }

    // 5. four conv + BN + leaky blocks (ping-pong bufA/bufB)
    unsigned int* inb  = bufA;
    unsigned int* outb = bufB;
    for (int i = 0; i < 4; ++i) {
        int hin = Hin[i], win = Hin[i];
        int ho  = hin - 2, wo = win - 2;
        int HW  = ho * wo;

        conv3x3_wmma_bf16<<<dim3((wo + 127) / 128, ho, Cout[i] / 32), dim3(256), 0, stream>>>(
            inb, wpk[i], outb, CinPad[i], Cout[i], hin, win);

        stats_kernel<<<dim3((HW + 4095) / 4096), dim3(256), 0, stream>>>(
            outb, mbuf[i], &sum[statOff[i]], &sq[statOff[i]], HW, lgC[i]);

        finalize_kernel<<<dim3((Cout[i] + 255) / 256), dim3(256), 0, stream>>>(
            &sum[statOff[i]], &sq[statOff[i]], &counts[i], g[i], be[i],
            &scale[statOff[i]], &shift[statOff[i]], Cout[i]);

        int total = (Cout[i] / 2) * HW;
        norm_kernel<<<dim3((total + 255) / 256), dim3(256), 0, stream>>>(
            outb, mbuf[i], &scale[statOff[i]], &shift[statOff[i]], lgC[i], total);

        unsigned int* t = inb; inb = outb; outb = t;
    }

    // 6. tail conv 32->1 + final mask multiply -> d_out (fp32 758x758)
    {
        int hin = 760, ho = 758;
        tail_conv_kernel<<<dim3((ho + 15) / 16, (ho + 15) / 16), dim3(256), 0, stream>>>(
            inb, wt, bt, mbuf[4], outp, hin, hin);
    }
}
